// FP8Linear_12704513261794
// MI455X (gfx1250) — 8-bit matrix operand order repaired and checked by reading, compiled, not run on hardware
//
#include <hip/hip_runtime.h>

typedef __attribute__((ext_vector_type(16))) int          v16i;
typedef __attribute__((ext_vector_type(8)))  float        v8f;
typedef __attribute__((ext_vector_type(8)))  int          v8i_;
typedef __attribute__((ext_vector_type(4)))  int          v4i;
typedef __attribute__((ext_vector_type(4)))  unsigned int v4u;

#define M_DIM  16384
#define N_DIM  4096
#define K_DIM  4096
#define BLK_M  128
#define BLK_N  128
#define BLK_K  128
#define F8_MAX 448.0f

__device__ __forceinline__ unsigned int f32_to_e4m3(float f) {
    f = fminf(fmaxf(f, -F8_MAX), F8_MAX);
    unsigned int u = __float_as_uint(f);
    unsigned int s = (u >> 24) & 0x80u;
    int e          = (int)((u >> 23) & 0xFFu) - 127;
    unsigned int m = u & 0x7FFFFFu;
    if (e >= -6) {
        unsigned int q   = ((unsigned int)(e + 7) << 3) | (m >> 20);
        unsigned int rem = m & 0xFFFFFu;
        q += (rem > 0x80000u) || ((rem == 0x80000u) && (q & 1u));
        if (q > 0x7Eu) q = 0x7Eu;
        return s | q;
    }
    if (e < -10) return s;
    unsigned int mant  = m | 0x800000u;
    int          rsh   = 20 + (-6 - e);
    unsigned int q     = mant >> rsh;
    unsigned int rem   = mant & ((1u << rsh) - 1u);
    unsigned int halfv = 1u << (rsh - 1);
    q += (rem > halfv) || ((rem == halfv) && (q & 1u));
    return s | q;
}

__global__ void zero_amax_kernel(unsigned int* ws) {
    if (threadIdx.x < 2) ws[threadIdx.x] = 0u;
}

__global__ void amax_kernel(const float* __restrict__ x, size_t n4,
                            unsigned int* __restrict__ out) {
    const float4* xv = (const float4*)x;
    float m = 0.0f;
    for (size_t i = (size_t)blockIdx.x * blockDim.x + threadIdx.x; i < n4;
         i += (size_t)gridDim.x * blockDim.x) {
        float4 v = xv[i];
        m = fmaxf(m, fmaxf(fmaxf(fabsf(v.x), fabsf(v.y)),
                           fmaxf(fabsf(v.z), fabsf(v.w))));
    }
    __shared__ float red[256];
    red[threadIdx.x] = m;
    __syncthreads();
    for (int s = 128; s > 0; s >>= 1) {
        if ((int)threadIdx.x < s)
            red[threadIdx.x] = fmaxf(red[threadIdx.x], red[threadIdx.x + s]);
        __syncthreads();
    }
    if (threadIdx.x == 0) atomicMax(out, __float_as_uint(red[0]));
}

__global__ void quantize_kernel(const float* __restrict__ x,
                                unsigned char* __restrict__ q, size_t n4,
                                const unsigned int* __restrict__ amax) {
    const float inv = F8_MAX / __uint_as_float(*amax);
    const float4* xv = (const float4*)x;
    unsigned int* qv = (unsigned int*)q;
    for (size_t i = (size_t)blockIdx.x * blockDim.x + threadIdx.x; i < n4;
         i += (size_t)gridDim.x * blockDim.x) {
        float4 v = xv[i];
        unsigned int b = f32_to_e4m3(v.x * inv)
                       | (f32_to_e4m3(v.y * inv) << 8)
                       | (f32_to_e4m3(v.z * inv) << 16)
                       | (f32_to_e4m3(v.w * inv) << 24);
        qv[i] = b;
    }
}

__device__ __forceinline__ void tdm_load_tile_2d(unsigned int lds_addr,
                                                 const unsigned char* gptr,
                                                 unsigned int tensor_d0,
                                                 unsigned int tensor_d1,
                                                 unsigned int stride0,
                                                 unsigned int tile_d0,
                                                 unsigned int tile_d1) {
    unsigned long long ga = (unsigned long long)(uintptr_t)gptr;
    v4u g0;
    g0[0] = 1u;
    g0[1] = lds_addr;
    g0[2] = (unsigned int)(ga & 0xFFFFFFFFu);
    g0[3] = (unsigned int)((ga >> 32) & 0x01FFFFFFu) | (2u << 30);
    v8i_ g1;
    g1[0] = 0;
    g1[1] = (int)((tensor_d0 & 0xFFFFu) << 16);
    g1[2] = (int)(((tensor_d0 >> 16) & 0xFFFFu) |
                  ((tensor_d1 & 0xFFFFu) << 16));
    g1[3] = (int)(((tensor_d1 >> 16) & 0xFFFFu) |
                  ((tile_d0 & 0xFFFFu) << 16));
    g1[4] = (int)(tile_d1 & 0xFFFFu);
    g1[5] = (int)stride0;
    g1[6] = 0;
    g1[7] = 0;
    v4i z4 = {0, 0, 0, 0};
#if __has_include(<hip/amd_detail/amd_gfx1250_TDM.h>)
    v8i_ z8 = {0, 0, 0, 0, 0, 0, 0, 0};
    __builtin_amdgcn_tensor_load_to_lds(g0, g1, z4, z4, z8, 0);
#else
    __builtin_amdgcn_tensor_load_to_lds(g0, g1, z4, z4, 0);
#endif
}

__global__ __launch_bounds__(128)
void fp8_gemm_kernel(const unsigned char* __restrict__ Aq,
                     const unsigned char* __restrict__ Bq,
                     const float* __restrict__ bias,
                     const unsigned int* __restrict__ amax,
                     float* __restrict__ out) {
    __shared__ __align__(16) unsigned char lA[2][BLK_M * BLK_K];
    __shared__ __align__(16) unsigned char lB[2][BLK_N * BLK_K];

    const int tid  = threadIdx.x;
    const int wave = tid >> 5;
    const int lane = tid & 31;
    const int half = lane >> 4;
    const int l16  = lane & 15;

    const int m0 = blockIdx.y * BLK_M;
    const int n0 = blockIdx.x * BLK_N;
    const int wm = (wave & 1) * 64;
    const int wn = (wave >> 1) * 64;

    const unsigned int ldsA = (unsigned int)(uintptr_t)(void*)&lA[0][0];
    const unsigned int ldsB = (unsigned int)(uintptr_t)(void*)&lB[0][0];
    const unsigned char* gA = Aq + (size_t)m0 * K_DIM;
    const unsigned char* gB = Bq + (size_t)n0 * K_DIM;

    v8f acc[4][4] = {};

    if (wave == 0) {
        tdm_load_tile_2d(ldsA, gA, K_DIM, M_DIM, K_DIM, BLK_K, BLK_M);
        tdm_load_tile_2d(ldsB, gB, K_DIM, N_DIM, K_DIM, BLK_K, BLK_N);
    }

    int buf = 0;
    const int KSTEPS = K_DIM / BLK_K;
    for (int ks = 0; ks < KSTEPS; ++ks) {
        if (wave == 0) __builtin_amdgcn_s_wait_tensorcnt(0);
        __syncthreads();

        const bool pf = (ks + 1) < KSTEPS;
        if (pf && wave == 0) {
            const int k0 = (ks + 1) * BLK_K;
            const unsigned int off = (unsigned int)((buf ^ 1) * BLK_M * BLK_K);
            tdm_load_tile_2d(ldsA + off, gA + k0, K_DIM, M_DIM, K_DIM,
                             BLK_K, BLK_M);
            tdm_load_tile_2d(ldsB + off, gB + k0, K_DIM, N_DIM, K_DIM,
                             BLK_K, BLK_N);
        }

        union { v16i v; unsigned long long u[8]; } afr[4];
#pragma unroll
        for (int i = 0; i < 4; ++i) {
            const unsigned long long* p = (const unsigned long long*)
                &lA[buf][(wm + i * 16 + l16) * BLK_K];
#pragma unroll
            for (int c = 0; c < 8; ++c) afr[i].u[c] = p[(c >> 1) * 4 + half * 2 + (c & 1)];
        }
        union { v16i v; uint4 q[4]; } bfr[4];
#pragma unroll
        for (int j = 0; j < 4; ++j) {
            const uint4* p = (const uint4*)
                &lB[buf][(wn + j * 16 + l16) * BLK_K];
#pragma unroll
            for (int c = 0; c < 4; ++c) bfr[j].q[c] = p[c * 2 + half];
        }

#pragma unroll
        for (int i = 0; i < 4; ++i)
#pragma unroll
            for (int j = 0; j < 4; ++j)
                acc[i][j] = __builtin_amdgcn_wmma_f32_16x16x128_fp8_fp8(
                    afr[i].v, bfr[j].v, (short)0, acc[i][j],
                     false,  false);

        if (pf) buf ^= 1;
    }

    const float scale = (__uint_as_float(amax[0]) * (1.0f / F8_MAX)) *
                        (__uint_as_float(amax[1]) * (1.0f / F8_MAX));
#pragma unroll
    for (int j = 0; j < 4; ++j) {
        const int col = n0 + wn + j * 16 + l16;
        const float bv = bias[col];
#pragma unroll
        for (int i = 0; i < 4; ++i) {
            const int rbase = m0 + wm + i * 16 + half * 8;
            float* op = &out[(size_t)rbase * N_DIM + col];
#pragma unroll
            for (int r = 0; r < 8; ++r)
                op[(size_t)r * N_DIM] = acc[i][j][r] * scale + bv;
        }
    }
}

extern "C" void kernel_launch(void* const* d_in, const int* in_sizes, int n_in,
                              void* d_out, int out_size, void* d_ws,
                              size_t ws_size, hipStream_t stream) {
    const float* x    = (const float*)d_in[0];
    const float* w    = (const float*)d_in[1];
    const float* bias = (const float*)d_in[2];
    float* out        = (float*)d_out;

    unsigned int*  amax = (unsigned int*)d_ws;
    unsigned char* xq   = (unsigned char*)d_ws + 256;
    unsigned char* wq   = xq + (size_t)M_DIM * K_DIM;

    const size_t nx4 = (size_t)M_DIM * K_DIM / 4;
    const size_t nw4 = (size_t)N_DIM * K_DIM / 4;

    zero_amax_kernel<<<1, 64, 0, stream>>>(amax);
    amax_kernel<<<2048, 256, 0, stream>>>(x, nx4, amax + 0);
    amax_kernel<<<1024, 256, 0, stream>>>(w, nw4, amax + 1);
    quantize_kernel<<<4096, 256, 0, stream>>>(x, xq, nx4, amax + 0);
    quantize_kernel<<<2048, 256, 0, stream>>>(w, wq, nw4, amax + 1);

    dim3 grid(N_DIM / BLK_N, M_DIM / BLK_M);
    fp8_gemm_kernel<<<grid, 128, 0, stream>>>(xq, wq, bias, amax, out);
}
